// GCNGraphDTA_73882027425856
// MI455X (gfx1250) — matrix operand order repaired and checked by reading, not compiled, not run on hardware
//
#include <hip/hip_runtime.h>
#include <hip/hip_bf16.h>

#define N_NODES 51200
#define N_GRAPHS 2048
#define NPG 25
#define N_EDGES 819200
#define F_IN 13
#define HID 128
#define ECAP 1024            // LDS edge cache per graph (mean ~400, max ~470)
#define WPK_PER_LAYER (8 * 4 * 32 * 16)   // nt*kt*lane*halves = 16384 halves

typedef __attribute__((ext_vector_type(16))) _Float16 v16h;
typedef __attribute__((ext_vector_type(8)))  float    v8f;

// ---------------- edge bucketing ----------------

__global__ void zero_cnt_kernel(int* __restrict__ cnt, int n) {
    int i = blockIdx.x * blockDim.x + threadIdx.x;
    if (i < n) cnt[i] = 0;
}

__global__ void count_kernel(const int* __restrict__ dst, int* __restrict__ cnt, int nE) {
    int e = blockIdx.x * blockDim.x + threadIdx.x;
    if (e < nE) atomicAdd(&cnt[dst[e] / NPG], 1);
}

// exclusive scan of 2048 counts with 256 threads x 8 elems each
__global__ void scan_kernel(const int* __restrict__ cnt, int* __restrict__ off,
                            int* __restrict__ pos) {
    __shared__ int part[256];
    int t = threadIdx.x;
    int local[8];
    int s = 0;
#pragma unroll
    for (int i = 0; i < 8; ++i) { local[i] = s; s += cnt[t * 8 + i]; }
    part[t] = s;
    __syncthreads();
    for (int d = 1; d < 256; d <<= 1) {
        int v = part[t];
        int add = (t >= d) ? part[t - d] : 0;
        __syncthreads();
        part[t] = v + add;
        __syncthreads();
    }
    int base = (t == 0) ? 0 : part[t - 1];
#pragma unroll
    for (int i = 0; i < 8; ++i) {
        int o = base + local[i];
        off[t * 8 + i] = o;
        pos[t * 8 + i] = o;
    }
    if (t == 255) off[2048] = part[255];
}

__global__ void scatter_kernel(const int* __restrict__ src, const int* __restrict__ dst,
                               int* __restrict__ pos, int* __restrict__ epk, int nE) {
    int e = blockIdx.x * blockDim.x + threadIdx.x;
    if (e < nE) {
        int d = dst[e];
        int g = d / NPG;
        int p = atomicAdd(&pos[g], 1);
        int ls = src[e] - g * NPG;   // 0..24
        int ld = d - g * NPG;        // 0..24
        epk[p] = ls | (ld << 5);
    }
}

// ---------------- weight fragment pre-pack (f16, ISA 7.12.2 B layout) ----------------
// Wpk[L][((nt*4+kt)*32+lane)*16 + j] = f16( W[k*128+n] ), k = kt*32 + j + 8*(j>=8) + 8*(lane>=16),
// n = nt*16 + (lane&15); zero-padded past kdim (layer 1: kdim=13).
__global__ void pack_w_kernel(const float* __restrict__ W1, const float* __restrict__ W2,
                              const float* __restrict__ W3, _Float16* __restrict__ Wpk) {
    int idx = blockIdx.x * blockDim.x + threadIdx.x;
    if (idx >= 3 * WPK_PER_LAYER) return;
    int L = idx / WPK_PER_LAYER;
    int r = idx - L * WPK_PER_LAYER;
    int j    = r & 15;
    int lane = (r >> 4) & 31;
    int kt   = (r >> 9) & 3;
    int nt   = (r >> 11) & 7;
    int k = kt * 32 + j + ((j & 8) ? 8 : 0) + ((lane >= 16) ? 8 : 0);
    int n = nt * 16 + (lane & 15);
    const float* W = (L == 0) ? W1 : (L == 1) ? W2 : W3;
    int kdim = (L == 0) ? F_IN : HID;
    Wpk[idx] = (_Float16)((k < kdim) ? W[k * 128 + n] : 0.0f);
}

// ---------------- WMMA fragment helpers ----------------

// A fragment 16x32 f16 from LDS f32 (pitch 128): lane -> m = mbase + (lane&15);
// half j -> k = kb + j + (j&8?8:0) + (lane>=16?8:0)
__device__ __forceinline__ v16h frag_A(const float* __restrict__ Hs, int mbase,
                                       int kb, int kdim, int lane) {
    int m = mbase + (lane & 15);
    int khi = (lane >= 16) ? 8 : 0;
    v16h a;
#pragma unroll
    for (int j = 0; j < 16; ++j) {
        int k = kb + j + ((j & 8) ? 8 : 0) + khi;
        a[j] = (_Float16)((k < kdim) ? Hs[m * 128 + k] : 0.0f);
    }
    return a;
}

#define WMMA_F16(a, b, c) \
    __builtin_amdgcn_wmma_f32_16x16x32_f16(false, (a), false, (b), (short)0, (c), false, false)

// 32xK (LDS f32) @ KxHID (pre-packed f16 fragments) -> HWs (LDS f32 32x128)
__device__ __forceinline__ void gemm_wmma(const float* __restrict__ Hs,
                                          float* __restrict__ HWs,
                                          const _Float16* __restrict__ Wpk,
                                          int kdim, int ksteps, int wave, int lane) {
    v8f acc00 = {}, acc01 = {}, acc10 = {}, acc11 = {};
    int nt0 = 2 * wave, nt1 = nt0 + 1;
    for (int kt = 0; kt < ksteps; ++kt) {
        int kb = kt * 32;
        v16h a0 = frag_A(Hs, 0, kb, kdim, lane);
        v16h a1 = frag_A(Hs, 16, kb, kdim, lane);
        v16h b0 = *(const v16h*)(Wpk + ((nt0 * 4 + kt) * 32 + lane) * 16);
        v16h b1 = *(const v16h*)(Wpk + ((nt1 * 4 + kt) * 32 + lane) * 16);
        acc00 = WMMA_F16(a0, b0, acc00);
        acc01 = WMMA_F16(a0, b1, acc01);
        acc10 = WMMA_F16(a1, b0, acc10);
        acc11 = WMMA_F16(a1, b1, acc11);
    }
    // D layout: c[r] -> m = r + 8*(lane>=16), n = lane&15
    int hi = (lane >> 4) & 1;
    int c0 = nt0 * 16 + (lane & 15);
    int c1 = nt1 * 16 + (lane & 15);
#pragma unroll
    for (int r = 0; r < 8; ++r) {
        int m = r + 8 * hi;
        HWs[m * 128 + c0] = acc00[r];
        HWs[m * 128 + c1] = acc01[r];
        HWs[(16 + m) * 128 + c0] = acc10[r];
        HWs[(16 + m) * 128 + c1] = acc11[r];
    }
}

// column-parallel aggregation: thread c owns feature column c (no races)
__device__ __forceinline__ void aggregate(float* __restrict__ Hs,
                                          const float* __restrict__ HWs,
                                          const float* __restrict__ bias,
                                          const float* __restrict__ dinv_s,
                                          const float* __restrict__ dinv2_s,
                                          const int* __restrict__ eps_s, int nle,
                                          const int* __restrict__ ep_g, int ne, int c) {
    float bc = bias[c];
#pragma unroll
    for (int i = 0; i < NPG; ++i)
        Hs[i * 128 + c] = HWs[i * 128 + c] * dinv2_s[i] + bc;   // self-loop term + bias
    for (int e = 0; e < nle; ++e) {                             // LDS-cached edges
        int p = eps_s[e];
        int ls = p & 31, ld = (p >> 5) & 31;
        Hs[ld * 128 + c] += dinv_s[ls] * dinv_s[ld] * HWs[ls * 128 + c];
    }
    for (int e = nle; e < ne; ++e) {                            // (rare) overflow tail
        int p = ep_g[e];
        int ls = p & 31, ld = (p >> 5) & 31;
        Hs[ld * 128 + c] += dinv_s[ls] * dinv_s[ld] * HWs[ls * 128 + c];
    }
#pragma unroll
    for (int i = 0; i < NPG; ++i)
        Hs[i * 128 + c] = fmaxf(Hs[i * 128 + c], 0.0f);         // relu
}

// ---------------- fused per-graph kernel ----------------

__global__ void __launch_bounds__(128) gcn_fused_kernel(
        const float* __restrict__ x, const float* __restrict__ prot,
        const _Float16* __restrict__ Wpk,
        const float* __restrict__ b1, const float* __restrict__ b2,
        const float* __restrict__ b3,
        const float* __restrict__ Wf1, const float* __restrict__ bf1,
        const float* __restrict__ Wf2, const float* __restrict__ bf2,
        const int* __restrict__ e_off, const int* __restrict__ e_pk,
        float* __restrict__ out) {
    __shared__ float Hs[32 * 128];
    __shared__ float HWs[32 * 128];
    __shared__ float dinv_s[NPG];
    __shared__ float dinv2_s[NPG];
    __shared__ float comb_s[256];
    __shared__ float red_s[128];
    __shared__ int   eps_s[ECAP];

    int g = blockIdx.x;
    int tid = threadIdx.x;
    int lane = tid & 31;
    int wave = tid >> 5;

    int e_beg = e_off[g];
    int ne = e_off[g + 1] - e_beg;
    const int* ep = e_pk + e_beg;
    int nle = (ne < ECAP) ? ne : ECAP;

    // zero H (rows 25..31 stay zero forever; cols 13..127 zero for layer 1)
    for (int i = tid; i < 32 * 128; i += 128) Hs[i] = 0.0f;
    // stage this graph's edge list into LDS with gfx1250 async-to-LDS DMA
    for (int i = tid; i < nle; i += 128) {
        unsigned lds_off = (unsigned)(unsigned long long)&eps_s[i];
        asm volatile("global_load_async_to_lds_b32 %0, %1, %2"
                     :: "v"(lds_off), "v"(i * 4), "s"(ep) : "memory");
    }
    asm volatile("s_wait_asynccnt 0x0" ::: "memory");
    __syncthreads();

    // load node features (25 x 13)
    for (int i = tid; i < NPG * F_IN; i += 128) {
        int r = i / F_IN, c = i - r * F_IN;
        Hs[r * 128 + c] = x[(g * NPG + r) * F_IN + c];
    }
    // per-graph degrees + dinv (includes self-loop)
    if (tid < NPG) {
        float d = 1.0f;
        for (int e = 0; e < nle; ++e)
            d += (((eps_s[e] >> 5) & 31) == tid) ? 1.0f : 0.0f;
        for (int e = nle; e < ne; ++e)
            d += (((ep[e] >> 5) & 31) == tid) ? 1.0f : 0.0f;
        float dv = rsqrtf(d);
        dinv_s[tid] = dv;
        dinv2_s[tid] = dv * dv;
    }
    __syncthreads();

    // layer 1: K = 13 (one padded K-step)
    gemm_wmma(Hs, HWs, Wpk, F_IN, 1, wave, lane);
    __syncthreads();
    aggregate(Hs, HWs, b1, dinv_s, dinv2_s, eps_s, nle, ep, ne, tid);
    __syncthreads();
    // layer 2
    gemm_wmma(Hs, HWs, Wpk + WPK_PER_LAYER, HID, 4, wave, lane);
    __syncthreads();
    aggregate(Hs, HWs, b2, dinv_s, dinv2_s, eps_s, nle, ep, ne, tid);
    __syncthreads();
    // layer 3
    gemm_wmma(Hs, HWs, Wpk + 2 * WPK_PER_LAYER, HID, 4, wave, lane);
    __syncthreads();
    aggregate(Hs, HWs, b3, dinv_s, dinv2_s, eps_s, nle, ep, ne, tid);
    __syncthreads();

    // global max pool (over 25 real rows) + concat prot
    {
        float m = Hs[tid];
#pragma unroll
        for (int i = 1; i < NPG; ++i) m = fmaxf(m, Hs[i * 128 + tid]);
        comb_s[tid] = m;
        comb_s[128 + tid] = prot[g * 128 + tid];
    }
    __syncthreads();

    // MLP head: thread j computes outputs j and j+128 of FC1, then dot with Wf2
    float o0 = bf1[tid], o1 = bf1[tid + 128];
    for (int k = 0; k < 256; ++k) {
        float cv = comb_s[k];
        o0 = fmaf(cv, Wf1[k * 256 + tid], o0);
        o1 = fmaf(cv, Wf1[k * 256 + tid + 128], o1);
    }
    float s = fmaxf(o0, 0.0f) * Wf2[tid] + fmaxf(o1, 0.0f) * Wf2[tid + 128];
    red_s[tid] = s;
    __syncthreads();
    for (int o = 64; o > 0; o >>= 1) {
        if (tid < o) red_s[tid] += red_s[tid + o];
        __syncthreads();
    }
    if (tid == 0) out[g] = red_s[0] + bf2[0];
}

// ---------------- launch ----------------

extern "C" void kernel_launch(void* const* d_in, const int* in_sizes, int n_in,
                              void* d_out, int out_size, void* d_ws, size_t ws_size,
                              hipStream_t stream) {
    const float* x    = (const float*)d_in[0];
    const int*   ei   = (const int*)d_in[1];
    // d_in[2] = batch (unused: batch id == node/25)
    const float* prot = (const float*)d_in[3];
    const float* W1 = (const float*)d_in[4];
    const float* b1 = (const float*)d_in[5];
    const float* W2 = (const float*)d_in[6];
    const float* b2 = (const float*)d_in[7];
    const float* W3 = (const float*)d_in[8];
    const float* b3 = (const float*)d_in[9];
    const float* Wf1 = (const float*)d_in[10];
    const float* bf1 = (const float*)d_in[11];
    const float* Wf2 = (const float*)d_in[12];
    const float* bf2 = (const float*)d_in[13];

    const int* src = ei;
    const int* dst = ei + N_EDGES;

    // workspace: [packed f16 weight fragments][edge bucket arrays]
    _Float16* Wpk = (_Float16*)d_ws;                         // 3*16384 halves = 96 KB
    int* e_cnt = (int*)((char*)d_ws + 3 * WPK_PER_LAYER * sizeof(_Float16));
    int* e_off = e_cnt + N_GRAPHS;                           // [2049]
    int* e_pos = e_off + (N_GRAPHS + 1);                     // [2048]
    int* e_pk  = e_pos + N_GRAPHS;                           // [819200]

    pack_w_kernel<<<(3 * WPK_PER_LAYER + 255) / 256, 256, 0, stream>>>(W1, W2, W3, Wpk);
    zero_cnt_kernel<<<(N_GRAPHS + 255) / 256, 256, 0, stream>>>(e_cnt, N_GRAPHS);
    count_kernel<<<(N_EDGES + 255) / 256, 256, 0, stream>>>(dst, e_cnt, N_EDGES);
    scan_kernel<<<1, 256, 0, stream>>>(e_cnt, e_off, e_pos);
    scatter_kernel<<<(N_EDGES + 255) / 256, 256, 0, stream>>>(src, dst, e_pos, e_pk, N_EDGES);

    gcn_fused_kernel<<<N_GRAPHS, 128, 0, stream>>>(
        x, prot, Wpk, b1, b2, b3, Wf1, bf1, Wf2, bf2,
        e_off, e_pk, (float*)d_out);
}
